// Classifier_3813930959544
// MI455X (gfx1250) — matrix operand order repaired and checked by reading, not compiled, not run on hardware
//
#include <hip/hip_runtime.h>

typedef _Float16 v16h __attribute__((ext_vector_type(16)));
typedef _Float16 v8h  __attribute__((ext_vector_type(8)));
typedef float    v8f  __attribute__((ext_vector_type(8)));

#define B_ROWS 16384
#define G_ROWS 8192
#define DDIM   64

#define LOG2E      1.4426950408889634f
#define NHALFLOG2E (-0.72134752044448170f)   // -0.5 * log2(e)

// ---------------------------------------------------------------------------
// Kernel 1: one wave32 per row. Convert f32 -> f16 (for WMMA) and:
//   B rows:  nA[row] = -0.5*log2(e) * ||x||^2          (stays in exp arg)
//   G rows:  w2[row] = w[row] * 2^(-0.5*log2(e)*||c||^2)  (norm folded into
//            the weight so the main loop's exp arg needs no center norm)
// ---------------------------------------------------------------------------
__global__ void __launch_bounds__(256)
prep_kernel(const float* __restrict__ inA, const float* __restrict__ inG,
            const float* __restrict__ wgt,
            _Float16* __restrict__ Ah, _Float16* __restrict__ Gh,
            float* __restrict__ nA, float* __restrict__ w2) {
    const int wave = (blockIdx.x * blockDim.x + threadIdx.x) >> 5;
    const int lane = threadIdx.x & 31;
    if (wave >= B_ROWS + G_ROWS) return;

    const bool isB = (wave < B_ROWS);
    const int  row = isB ? wave : (wave - B_ROWS);
    const float* src = isB ? inA : inG;
    _Float16*    dst = isB ? Ah  : Gh;

    const float2 v = *(const float2*)(src + (size_t)row * DDIM + lane * 2);
    float s = v.x * v.x + v.y * v.y;
    dst[(size_t)row * DDIM + lane * 2 + 0] = (_Float16)v.x;
    dst[(size_t)row * DDIM + lane * 2 + 1] = (_Float16)v.y;
#pragma unroll
    for (int m = 16; m >= 1; m >>= 1) s += __shfl_xor(s, m, 32);
    if (lane == 0) {
        const float ns = NHALFLOG2E * s;
        if (isB) nA[row] = ns;
        else     w2[row] = wgt[row] * __builtin_amdgcn_exp2f(ns);
    }
}

// ---------------------------------------------------------------------------
// Kernel 2: fused GEMM + Gaussian kernel + weighted reduction.
// Block = 256 threads (8 waves). Block owns a 16-row slab of B.
// Wave w processes G-tiles w, w+8, ... (64 tiles of 16 centers, K=64 via two
// v_wmma_f32_16x16x32_f16). Per element:
//   acc += w2[g] * exp2( log2(e)*dot + nbs[m] )     -- 2 fma + 1 v_exp_f32
// Butterfly shuffles + LDS ds_add_f32 finish the G-sum across waves.
// ---------------------------------------------------------------------------
__global__ void __launch_bounds__(256)
rbf_kernel(const _Float16* __restrict__ Ah, const _Float16* __restrict__ Gh,
           const float* __restrict__ nA, const float* __restrict__ w2,
           float* __restrict__ out) {
    __shared__ float s[16];
    const int tid  = threadIdx.x;
    const int wave = tid >> 5;          // 0..7
    const int lane = tid & 31;
    const int hi   = lane >> 4;         // 0/1 half-wave
    const int col  = lane & 15;         // matrix row/col index within tile
    const int m0   = blockIdx.x * 16;   // slab base in B

    if (tid < 16) s[tid] = 0.0f;
    __syncthreads();

    // --- A fragments: 16-bit A 16x32 layout (contiguous K, matching B).
    // lanes 0-15 (hi=0): VGPR0-7 = K 0-15
    // lanes 16-31 (hi=1): VGPR0-7 = K 16-31
    v16h a0, a1;
    {
        const _Float16* arow = Ah + (size_t)(m0 + col) * DDIM;
        v8h lo0 = *(const v8h*)(arow + hi * 16);
        v8h up0 = *(const v8h*)(arow + hi * 16 + 8);
        v8h lo1 = *(const v8h*)(arow + 32 + hi * 16);
        v8h up1 = *(const v8h*)(arow + 32 + hi * 16 + 8);
#pragma unroll
        for (int i = 0; i < 8; ++i) {
            a0[i] = lo0[i]; a0[i + 8] = up0[i];
            a1[i] = lo1[i]; a1[i + 8] = up1[i];
        }
    }

    // Loop-invariant pre-scaled row norms for this lane's 8 output rows
    // (C layout: VGPR r holds row hi*8 + r).
    float nbv[8];
#pragma unroll
    for (int r = 0; r < 8; ++r) nbv[r] = nA[m0 + hi * 8 + r];

    float acc[8];
#pragma unroll
    for (int r = 0; r < 8; ++r) acc[r] = 0.0f;

    const int tilesPerWave = (G_ROWS / 16) / 8;   // 64, exact
    for (int t = 0; t < tilesPerWave; ++t) {
        const int g0 = (t * 8 + wave) * 16;
        const _Float16* grow = Gh + (size_t)(g0 + col) * DDIM;

        // B fragments: 16-bit B 32x16 layout — lane group holds 16
        // contiguous K values of column `col`.
        v16h b0 = *(const v16h*)(grow + hi * 16);        // K = 0..31
        v16h b1 = *(const v16h*)(grow + 32 + hi * 16);   // K = 32..63

        const float wg = w2[g0 + col];   // weight with center norm folded in

        v8f c = {};
        c = __builtin_amdgcn_wmma_f32_16x16x32_f16(false, a0, false, b0,
                                                   (short)0, c, false, false);
        c = __builtin_amdgcn_wmma_f32_16x16x32_f16(false, a1, false, b1,
                                                   (short)0, c, false, false);

        if (t + 1 < tilesPerWave)           // wave-uniform branch
            __builtin_prefetch((const void*)(grow + 8 * 16 * DDIM), 0, 3);

#pragma unroll
        for (int r = 0; r < 8; ++r) {
            const float arg = __builtin_fmaf(c[r], LOG2E, nbv[r]);
            acc[r] = __builtin_fmaf(wg, __builtin_amdgcn_exp2f(arg), acc[r]);
        }
    }

    // Butterfly-sum the 16 columns of each half-wave (width-16 groups).
#pragma unroll
    for (int r = 0; r < 8; ++r) {
#pragma unroll
        for (int m = 8; m >= 1; m >>= 1) acc[r] += __shfl_xor(acc[r], m, 16);
    }
    // Merge the 8 waves' partials via LDS float atomics.
    if (col == 0) {
#pragma unroll
        for (int r = 0; r < 8; ++r) atomicAdd(&s[hi * 8 + r], acc[r]);
    }
    __syncthreads();
    if (tid < 16) out[m0 + tid] = s[tid];
}

// ---------------------------------------------------------------------------
extern "C" void kernel_launch(void* const* d_in, const int* in_sizes, int n_in,
                              void* d_out, int out_size, void* d_ws, size_t ws_size,
                              hipStream_t stream) {
    (void)in_sizes; (void)n_in; (void)out_size; (void)ws_size;
    const float* input   = (const float*)d_in[0];   // [B, 64] f32
    const float* centers = (const float*)d_in[1];   // [G, 64] f32
    const float* weights = (const float*)d_in[2];   // [G]     f32
    float* out = (float*)d_out;                     // [B]     f32

    char* ws = (char*)d_ws;
    _Float16* Ah = (_Float16*)ws;  ws += (size_t)B_ROWS * DDIM * sizeof(_Float16);
    _Float16* Gh = (_Float16*)ws;  ws += (size_t)G_ROWS * DDIM * sizeof(_Float16);
    float*    nA = (float*)ws;     ws += (size_t)B_ROWS * sizeof(float);
    float*    w2 = (float*)ws;

    const int R = B_ROWS + G_ROWS;                  // 24576 rows, 8 waves/block
    prep_kernel<<<R / 8, 256, 0, stream>>>(input, centers, weights, Ah, Gh, nA, w2);
    rbf_kernel<<<B_ROWS / 16, 256, 0, stream>>>(Ah, Gh, nA, w2, out);
}
